// GINNodeEmbedding_69973607186521
// MI455X (gfx1250) — matrix operand order repaired and checked by reading, not compiled, not run on hardware
//
#include <hip/hip_runtime.h>

// ---------------------------------------------------------------------------
// GIN (node + bond-angle) forward for MI455X / gfx1250.
// GEMMs run as bf16 V_WMMA_F32_16X16X32_BF16 with fp32 accumulate (K=128).
// Scatter-add via global_atomic_add_f32. wave32 layouts per CDNA5 ISA 7.12.2.
// ---------------------------------------------------------------------------

typedef __attribute__((ext_vector_type(16))) __bf16 v16bf;
typedef __attribute__((ext_vector_type(8)))  __bf16 v8bf;
typedef __attribute__((ext_vector_type(8)))  float  v8f;

#define D 128
#define LDSS 136   // padded LDS row stride (bf16 elems): 4-bank stride/row -> conflict-free b128

// ---------------- small elementwise / embedding kernels --------------------

__global__ __launch_bounds__(128) void k_atom_embed(
    const int* __restrict__ xa, const float* __restrict__ tab,
    float* __restrict__ h, int N) {
  __shared__ int sid[9];
  int n = blockIdx.x, d = threadIdx.x;
  if (n >= N) return;
  if (d < 9) sid[d] = xa[(size_t)n * 9 + d];
  __syncthreads();
  float acc = 0.f;
#pragma unroll
  for (int f = 0; f < 9; ++f)
    acc += tab[((size_t)f * 124 + sid[f]) * D + d];
  h[(size_t)n * D + d] = acc;
}

__global__ __launch_bounds__(256) void k_rbf(
    const float* __restrict__ x, float* __restrict__ out, int n, int K) {
  int idx = blockIdx.x * blockDim.x + threadIdx.x;
  if (idx >= n * K) return;
  int e = idx / K, k = idx - e * K;
  float dlt = x[e] - 0.1f * (float)k;
  out[idx] = __expf(-10.f * dlt * dlt);
}

// h_ba feature: bias + 3 bond-id embeddings + rbf(20) @ W
__global__ __launch_bounds__(128) void k_bond_feat(
    const float* __restrict__ rbf, const float* __restrict__ W,
    const float* __restrict__ b, const float* __restrict__ emb,
    const int* __restrict__ eai, float* __restrict__ out, int E) {
  __shared__ float srbf[20];
  __shared__ int sid[3];
  int e = blockIdx.x, d = threadIdx.x;
  if (e >= E) return;
  if (d < 20) srbf[d] = rbf[(size_t)e * 20 + d];
  if (d >= 32 && d < 35) sid[d - 32] = eai[(size_t)e * 3 + (d - 32)];
  __syncthreads();
  float acc = b[d];
#pragma unroll
  for (int f = 0; f < 3; ++f)
    acc += emb[((size_t)f * 11 + sid[f]) * D + d];
#pragma unroll
  for (int k = 0; k < 20; ++k)
    acc = fmaf(srbf[k], W[k * D + d], acc);
  out[(size_t)e * D + d] = acc;
}

// node messages: agg[dst] += relu(h[src] + h_ba[e]); 32 lanes/edge, float4
__global__ __launch_bounds__(256) void k_node_message(
    const float* __restrict__ h, const float* __restrict__ hba,
    const int* __restrict__ ei, float* __restrict__ agg, int E) {
  int idx = blockIdx.x * blockDim.x + threadIdx.x;
  int e = idx >> 5;
  if (e >= E) return;
  int c = (idx & 31) * 4;
  int src = ei[e], dst = ei[E + e];
  float4 hv = *(const float4*)(h + (size_t)src * D + c);
  float4 bv = *(const float4*)(hba + (size_t)e * D + c);
  float m0 = fmaxf(hv.x + bv.x, 0.f);
  float m1 = fmaxf(hv.y + bv.y, 0.f);
  float m2 = fmaxf(hv.z + bv.z, 0.f);
  float m3 = fmaxf(hv.w + bv.w, 0.f);
  float* p = agg + (size_t)dst * D + c;
  atomicAdd(p + 0, m0);
  atomicAdd(p + 1, m1);
  atomicAdd(p + 2, m2);
  atomicAdd(p + 3, m3);
}

// angle messages with fused angle-RBF projection (avoids 205MB cur_ang buffer):
// agg_e[dst] += relu(cur_h_ba[src] + rbf_ang(e) @ ang_W + ang_b)
__global__ __launch_bounds__(128) void k_edge_message(
    const float* __restrict__ curhba, const float* __restrict__ rbf,
    const float* __restrict__ angW, const float* __restrict__ angb,
    const int* __restrict__ eiba, float* __restrict__ aggE, int EBA) {
  __shared__ float srbf[32];
  int e = blockIdx.x, d = threadIdx.x;
  if (e >= EBA) return;
  if (d < 32) srbf[d] = rbf[(size_t)e * 32 + d];
  __syncthreads();
  int src = eiba[e], dst = eiba[EBA + e];
  float acc = angb[d];
#pragma unroll
  for (int k = 0; k < 32; ++k)
    acc = fmaf(srbf[k], angW[k * D + d], acc);
  float m = fmaxf(curhba[(size_t)src * D + d] + acc, 0.f);
  atomicAdd(&aggE[(size_t)dst * D + d], m);
}

// ---------------- fused GIN MLP: WMMA bf16, fp32 accumulate ----------------
// y = [relu]( BN( ((1+eps)x + agg) @ W1 + b1 ) ) relu @ W2 + b2
// block = 128 thr = 4 waves; wave owns 16 rows x 128 cols.

// A fragment, 16-bit 16x32 layout (ISA 7.12.2): lane<16 holds K{0..7,16..23},
// lane>=16 holds K{8..15,24..31} of row (lane&15). Two 16B chunks, 16 apart.
static __device__ __forceinline__ v16bf ld_frag_split(const __bf16* s) {
  v8bf lo = *(const v8bf*)(s);
  v8bf hi = *(const v8bf*)(s + 16);
  v16bf a;
#pragma unroll
  for (int i = 0; i < 8; ++i) { a[i] = lo[i]; a[i + 8] = hi[i]; }
  return a;
}
// B fragment, contiguous k order (A uses the same): lane<16 = col lane, K 0..15 contiguous;
// lane>=16 = col lane-16, K 16..31. One contiguous 32B chunk (W stored K-minor).
static __device__ __forceinline__ v16bf ld_frag_contig(const __bf16* s) {
  v8bf lo = *(const v8bf*)(s);
  v8bf hi = *(const v8bf*)(s + 8);
  v16bf b;
#pragma unroll
  for (int i = 0; i < 8; ++i) { b[i] = lo[i]; b[i + 8] = hi[i]; }
  return b;
}

__global__ __launch_bounds__(128) void k_gin_mlp(
    const float* __restrict__ x, const float* __restrict__ agg,
    const float* __restrict__ eps_ptr,
    const float* __restrict__ W1, const float* __restrict__ b1,
    const float* __restrict__ g, const float* __restrict__ be,
    const float* __restrict__ rm, const float* __restrict__ rv,
    const float* __restrict__ W2, const float* __restrict__ b2,
    float* __restrict__ out, int M, int final_relu) {
  extern __shared__ __bf16 smem[];
  __bf16* sW1 = smem;                 // [128][LDSS]  W1^T (n-major, K contiguous)
  __bf16* sW2 = sW1 + 128 * LDSS;     // [128][LDSS]  W2^T
  __bf16* sX  = sW2 + 128 * LDSS;     // 4 waves x [16][LDSS] row tiles
  int tid = threadIdx.x;

  // Weights -> LDS transposed bf16 (thread t owns output column t).
  for (int k = 0; k < 128; ++k) {
    sW1[tid * LDSS + k] = (__bf16)W1[k * D + tid];
    sW2[tid * LDSS + k] = (__bf16)W2[k * D + tid];
  }
  __syncthreads();

  int wv = tid >> 5, lane = tid & 31;
  int hh = lane >> 4, nl = lane & 15;
  int r0 = blockIdx.x * 64 + wv * 16;
  __bf16* sXw = sX + wv * (16 * LDSS);
  float eps1 = 1.0f + eps_ptr[0];

  // Stage (1+eps)*x + agg as bf16 row tile.
#pragma unroll 4
  for (int i = 0; i < 16; ++i) {
    int gr = r0 + i;
    float4 xv = make_float4(0.f, 0.f, 0.f, 0.f), av = xv;
    if (gr < M) {
      xv = *(const float4*)(x + (size_t)gr * D + lane * 4);
      av = *(const float4*)(agg + (size_t)gr * D + lane * 4);
    }
    __bf16* dp = sXw + i * LDSS + lane * 4;
    dp[0] = (__bf16)fmaf(eps1, xv.x, av.x);
    dp[1] = (__bf16)fmaf(eps1, xv.y, av.y);
    dp[2] = (__bf16)fmaf(eps1, xv.z, av.z);
    dp[3] = (__bf16)fmaf(eps1, xv.w, av.w);
  }

  // ---- GEMM1: t = relu(BN(tile @ W1 + b1)) -> back into sXw as bf16 ----
  v16bf afr[4];
#pragma unroll
  for (int kt = 0; kt < 4; ++kt)
    afr[kt] = ld_frag_contig(sXw + nl * LDSS + kt * 32 + hh * 16);

#pragma unroll
  for (int nt = 0; nt < 8; ++nt) {
    v8f acc = {0.f, 0.f, 0.f, 0.f, 0.f, 0.f, 0.f, 0.f};
#pragma unroll
    for (int kt = 0; kt < 4; ++kt) {
      v16bf bf = ld_frag_contig(sW1 + (nt * 16 + nl) * LDSS + kt * 32 + hh * 16);
      acc = __builtin_amdgcn_wmma_f32_16x16x32_bf16(
          false, afr[kt], false, bf, (short)0, acc, false, false);
    }
    int n = nt * 16 + nl;
    float sc = g[n] * rsqrtf(rv[n] + 1e-5f);
    float sh = fmaf(b1[n] - rm[n], sc, be[n]);
#pragma unroll
    for (int r = 0; r < 8; ++r) {
      float z = fmaxf(fmaf(acc[r], sc, sh), 0.f);
      sXw[(r + hh * 8) * LDSS + n] = (__bf16)z;   // D-layout: VGPR r = row r(+8)
    }
  }

  // ---- GEMM2: out = t @ W2 + b2 (+relu) ----
#pragma unroll
  for (int kt = 0; kt < 4; ++kt)
    afr[kt] = ld_frag_contig(sXw + nl * LDSS + kt * 32 + hh * 16);

#pragma unroll
  for (int nt = 0; nt < 8; ++nt) {
    v8f acc = {0.f, 0.f, 0.f, 0.f, 0.f, 0.f, 0.f, 0.f};
#pragma unroll
    for (int kt = 0; kt < 4; ++kt) {
      v16bf bf = ld_frag_contig(sW2 + (nt * 16 + nl) * LDSS + kt * 32 + hh * 16);
      acc = __builtin_amdgcn_wmma_f32_16x16x32_bf16(
          false, afr[kt], false, bf, (short)0, acc, false, false);
    }
    int n = nt * 16 + nl;
    float bias = b2[n];
#pragma unroll
    for (int r = 0; r < 8; ++r) {
      float z = acc[r] + bias;
      if (final_relu) z = fmaxf(z, 0.f);
      int gr = r0 + r + hh * 8;
      if (gr < M) out[(size_t)gr * D + n] = z;
    }
  }
}

// --------------------------------- driver ----------------------------------

extern "C" void kernel_launch(void* const* d_in, const int* in_sizes, int n_in,
                              void* d_out, int out_size, void* d_ws, size_t ws_size,
                              hipStream_t stream) {
  (void)n_in; (void)out_size; (void)ws_size;
  const int N   = in_sizes[0] / 9;
  const int E   = in_sizes[1] / 2;
  const int EBA = in_sizes[4] / 2;
  const int L = 3;

  const int*   x_atom     = (const int*)d_in[0];
  const int*   edge_index = (const int*)d_in[1];
  const int*   eai        = (const int*)d_in[2];
  const float* eaf        = (const float*)d_in[3];
  const int*   ei_ba      = (const int*)d_in[4];
  const float* eab        = (const float*)d_in[5];
  const float* atom_emb   = (const float*)d_in[6];
  const float* bond_emb0  = (const float*)d_in[7];
  const float* bf_W0      = (const float*)d_in[8];
  const float* bf_b0      = (const float*)d_in[9];
  const float* bond_emb   = (const float*)d_in[10];
  const float* bf_W       = (const float*)d_in[11];
  const float* bf_b       = (const float*)d_in[12];
  const float* ang_W      = (const float*)d_in[13];
  const float* ang_b      = (const float*)d_in[14];
  const float* aW1 = (const float*)d_in[15]; const float* ab1 = (const float*)d_in[16];
  const float* ag  = (const float*)d_in[17]; const float* abe = (const float*)d_in[18];
  const float* arm = (const float*)d_in[19]; const float* arv = (const float*)d_in[20];
  const float* aW2 = (const float*)d_in[21]; const float* ab2 = (const float*)d_in[22];
  const float* aeps = (const float*)d_in[23];
  const float* bW1 = (const float*)d_in[24]; const float* bb1 = (const float*)d_in[25];
  const float* bg  = (const float*)d_in[26]; const float* bbe = (const float*)d_in[27];
  const float* brm = (const float*)d_in[28]; const float* brv = (const float*)d_in[29];
  const float* bW2 = (const float*)d_in[30]; const float* bb2 = (const float*)d_in[31];
  const float* beps = (const float*)d_in[32];

  float* out = (float*)d_out;

  // workspace carve
  char* w = (char*)d_ws; size_t off = 0;
  auto carve = [&](size_t bytes) -> float* {
    float* p = (float*)(w + off);
    off += (bytes + 255) & ~(size_t)255;
    return p;
  };
  float* rbf_bl  = carve((size_t)E * 20 * 4);
  float* rbf_ang = carve((size_t)EBA * 32 * 4);
  float* h       = carve((size_t)N * D * 4);
  float* hba     = carve((size_t)E * D * 4);
  float* aggN    = carve((size_t)N * D * 4);
  float* aggE    = carve((size_t)E * D * 4);
  float* curhba  = carve((size_t)E * D * 4);

  const size_t SMEM = (size_t)(2 * 128 + 4 * 16) * LDSS * sizeof(__bf16); // 87,040 B

  // initial features
  k_atom_embed<<<N, 128, 0, stream>>>(x_atom, atom_emb, h, N);
  k_rbf<<<(E * 20 + 255) / 256, 256, 0, stream>>>(eaf, rbf_bl, E, 20);
  k_rbf<<<(EBA * 32 + 255) / 256, 256, 0, stream>>>(eab, rbf_ang, EBA, 32);
  k_bond_feat<<<E, 128, 0, stream>>>(rbf_bl, bf_W0, bf_b0, bond_emb0, eai, hba, E);

  for (int l = 0; l < L; ++l) {
    const int last = (l == L - 1);
    // ---- node GIN ----
    hipMemsetAsync(aggN, 0, (size_t)N * D * 4, stream);
    k_node_message<<<((size_t)E * 32 + 255) / 256, 256, 0, stream>>>(
        h, hba, edge_index, aggN, E);
    float* nout = last ? out : h;  // in-place safe: block reads its rows first
    k_gin_mlp<<<(N + 63) / 64, 128, SMEM, stream>>>(
        h, aggN, aeps + l,
        aW1 + (size_t)l * D * D, ab1 + (size_t)l * D,
        ag + (size_t)l * D, abe + (size_t)l * D,
        arm + (size_t)l * D, arv + (size_t)l * D,
        aW2 + (size_t)l * D * D, ab2 + (size_t)l * D,
        nout, N, last ? 0 : 1);

    // ---- bond GIN ----
    k_bond_feat<<<E, 128, 0, stream>>>(
        rbf_bl, bf_W + (size_t)l * 20 * D, bf_b + (size_t)l * D,
        bond_emb + (size_t)l * 3 * 11 * D, eai, curhba, E);
    hipMemsetAsync(aggE, 0, (size_t)E * D * 4, stream);
    k_edge_message<<<EBA, 128, 0, stream>>>(
        curhba, rbf_ang, ang_W + (size_t)l * 32 * D, ang_b + (size_t)l * D,
        ei_ba, aggE, EBA);
    float* bout = last ? (out + (size_t)N * D) : hba;
    k_gin_mlp<<<(E + 63) / 64, 128, SMEM, stream>>>(
        curhba, aggE, beps + l,
        bW1 + (size_t)l * D * D, bb1 + (size_t)l * D,
        bg + (size_t)l * D, bbe + (size_t)l * D,
        brm + (size_t)l * D, brv + (size_t)l * D,
        bW2 + (size_t)l * D * D, bb2 + (size_t)l * D,
        bout, E, last ? 0 : 1);
  }
}
